// QuantumMotivicTile_56693568307585
// MI455X (gfx1250) — matrix operand order repaired and checked by reading, not compiled, not run on hardware
//
#include <hip/hip_runtime.h>
#include <hip/hip_bf16.h>

// ---------------------------------------------------------------------------
// Problem dims (fixed by the reference)
// ---------------------------------------------------------------------------
constexpr int Bz = 4;        // batch
constexpr int Sq = 2048;     // seq
constexpr int Dh = 512;      // hidden
constexpr int Ch = 8;        // cohomology dim
constexpr int Rk = 4;        // motive rank
constexpr int Mrows = Bz * Sq;        // 8192 rows in flattened GEMMs
constexpr float ATT_SCALE = 0.044194173824159216f; // 1/sqrt(512)

typedef __bf16 bf16_t;
typedef bf16_t v16bf __attribute__((ext_vector_type(16)));
typedef bf16_t v8bf  __attribute__((ext_vector_type(8)));
typedef float  v8f   __attribute__((ext_vector_type(8)));

// ---------------------------------------------------------------------------
// CDNA5 async global->LDS copy (ASYNCcnt-tracked).  16 bytes per lane.
// ---------------------------------------------------------------------------
__device__ __forceinline__ void qm_async_b128(unsigned int lds_off,
                                              unsigned int voff,
                                              const void* sbase) {
  asm volatile("global_load_async_to_lds_b128 %0, %1, %2"
               :: "v"(lds_off), "v"(voff), "s"(sbase)
               : "memory");
}
__device__ __forceinline__ void qm_wait_async0() {
  asm volatile("s_wait_asynccnt 0x0" ::: "memory");
}
__device__ __forceinline__ unsigned int qm_lds_off(const void* p) {
  // flat shared address: LDS byte offset lives in the low 32 bits (ISA 10.2)
  return (unsigned int)(unsigned long long)p;
}

// ---------------------------------------------------------------------------
// WMMA helpers (CDNA5 v_wmma_f32_16x16x32_bf16, wave32)
// ---------------------------------------------------------------------------
__device__ __forceinline__ v8f qm_wmma_bf16(v16bf a, v16bf b, v8f c) {
  return __builtin_amdgcn_wmma_f32_16x16x32_bf16(false, a, false, b, (short)0, c,
                                                 false, false);
}

// A-matrix 16x32 bf16 fragment: lane l (0..15) = row M=l,
// K 0..15; lane l+16 = same row, K 16..31 (matches contiguous B from LDS).
__device__ __forceinline__ v16bf qm_load_a(const bf16_t* base, int half) {
  v8bf lo = *(const v8bf*)(base + half * 16);
  v8bf hi = *(const v8bf*)(base + half * 16 + 8);
  v16bf r;
#pragma unroll
  for (int i = 0; i < 8; ++i) { r[i] = lo[i]; r[i + 8] = hi[i]; }
  return r;
}

// ---------------------------------------------------------------------------
// Kernel: complex -> planar bf16 (activations)
// ---------------------------------------------------------------------------
__global__ void qm_split_x(const float2* __restrict__ src,
                           bf16_t* __restrict__ re, bf16_t* __restrict__ im, int n) {
  int i = blockIdx.x * blockDim.x + threadIdx.x;
  if (i < n) {
    float2 v = src[i];
    re[i] = (bf16_t)v.x;
    im[i] = (bf16_t)v.y;
  }
}

// Kernel: complex weights -> planar bf16 (re, im, -im): no NEG needed in WMMA
__global__ void qm_split_w(const float2* __restrict__ src, bf16_t* __restrict__ wr,
                           bf16_t* __restrict__ wi, bf16_t* __restrict__ win, int n) {
  int i = blockIdx.x * blockDim.x + threadIdx.x;
  if (i < n) {
    float2 v = src[i];
    wr[i]  = (bf16_t)v.x;
    wi[i]  = (bf16_t)v.y;
    win[i] = (bf16_t)(-v.y);
  }
}

// ---------------------------------------------------------------------------
// Kernel: complex GEMM  out[m,n] = sum_k A[m,k]*W[n,k] + bias[n]
// Block = 8 waves, tile 256(M) x 64(N); each wave owns a 32(M)x64(N) complex
// strip (2 M-tiles x 4 N-tiles -> 32 WMMAs per K-step, 128 acc VGPRs).
// B chunk (64 rows x 32 K x 3 planes) double-buffered in LDS via
// global_load_async_to_lds_b128; B fragments from LDS feed 8 WMMAs each.
// Grid: (Mrows/256, Dh/64), block 256.
// ---------------------------------------------------------------------------
constexpr int SROW = 40;   // padded LDS row stride (32 data + 8 pad bf16)
constexpr int MT   = 2;    // M-tiles per wave

__global__ __launch_bounds__(256)
void qm_gemm_cplx(const bf16_t* __restrict__ Ar, const bf16_t* __restrict__ Ai,
                  const bf16_t* __restrict__ Wr, const bf16_t* __restrict__ Wi,
                  const bf16_t* __restrict__ Win, const float2* __restrict__ bias,
                  float2* __restrict__ out) {
  constexpr int K = Dh, N = Dh;
  constexpr int NKC = K / 32;                 // 16 K-chunks
  __shared__ bf16_t sB[2][3 * 64 * SROW];     // 2 x 15 KB

  const int tid  = threadIdx.x;
  const int wave = tid >> 5;
  const int lane = tid & 31;
  const int half = lane >> 4;
  const int l16  = lane & 15;

  const int mt0   = (blockIdx.x * 8 + wave) * MT;  // first 16-row tile of wave
  const int nbase = blockIdx.y * 64;

  const bf16_t* arp[MT];
  const bf16_t* aip[MT];
#pragma unroll
  for (int mt = 0; mt < MT; ++mt) {
    const int mrow = (mt0 + mt) * 16 + l16;
    arp[mt] = Ar + (size_t)mrow * K;
    aip[mt] = Ai + (size_t)mrow * K;
  }

  // async-copy role of this thread: B row (cn), 16B segment (cseg); one
  // transfer per plane (3 per thread per chunk) covers 64x32x3 bf16.
  const int cn   = tid >> 2;                  // 0..63
  const int cseg = tid & 3;                   // 0..3
  const unsigned int gbase =
      (unsigned int)(((nbase + cn) * K + cseg * 8) * (int)sizeof(bf16_t));
  const void* const wplane[3] = {Wr, Wi, Win};

  v8f accRe[MT][4], accIm[MT][4];
#pragma unroll
  for (int mt = 0; mt < MT; ++mt)
#pragma unroll
    for (int nt = 0; nt < 4; ++nt) {
      accRe[mt][nt] = v8f{0.f, 0.f, 0.f, 0.f, 0.f, 0.f, 0.f, 0.f};
      accIm[mt][nt] = v8f{0.f, 0.f, 0.f, 0.f, 0.f, 0.f, 0.f, 0.f};
    }

  // prologue: fill buffer 0
#pragma unroll
  for (int p = 0; p < 3; ++p) {
    qm_async_b128(qm_lds_off(&sB[0][(p * 64 + cn) * SROW + cseg * 8]),
                  gbase, wplane[p]);
  }

  for (int kc = 0; kc < NKC; ++kc) {
    qm_wait_async0();        // this wave's DMA into current buffer done
    __syncthreads();         // everyone's DMA done; prev-iter reads retired

    if (kc + 1 < NKC) {      // DMA of next chunk overlaps this chunk's math
      const unsigned int go = gbase + (unsigned int)((kc + 1) * 32 * sizeof(bf16_t));
      const int nb = (kc + 1) & 1;
#pragma unroll
      for (int p = 0; p < 3; ++p) {
        qm_async_b128(qm_lds_off(&sB[nb][(p * 64 + cn) * SROW + cseg * 8]),
                      go, wplane[p]);
      }
    }

    const bf16_t* sb = &sB[kc & 1][0];
    const int k0 = kc * 32;
    v16bf a_r[MT], a_i[MT];
#pragma unroll
    for (int mt = 0; mt < MT; ++mt) {
      a_r[mt] = qm_load_a(arp[mt] + k0, half);
      a_i[mt] = qm_load_a(aip[mt] + k0, half);
    }

#pragma unroll
    for (int nt = 0; nt < 4; ++nt) {
      const int nl = nt * 16 + l16;           // local B row
      v16bf b_r = *(const v16bf*)(sb + (0 * 64 + nl) * SROW + half * 16);
      v16bf b_i = *(const v16bf*)(sb + (1 * 64 + nl) * SROW + half * 16);
      v16bf b_n = *(const v16bf*)(sb + (2 * 64 + nl) * SROW + half * 16);
#pragma unroll
      for (int mt = 0; mt < MT; ++mt) {
        // Re += Ar*Wr + Ai*(-Wi);  Im += Ar*Wi + Ai*Wr
        accRe[mt][nt] = qm_wmma_bf16(a_r[mt], b_r, accRe[mt][nt]);
        accRe[mt][nt] = qm_wmma_bf16(a_i[mt], b_n, accRe[mt][nt]);
        accIm[mt][nt] = qm_wmma_bf16(a_r[mt], b_i, accIm[mt][nt]);
        accIm[mt][nt] = qm_wmma_bf16(a_i[mt], b_r, accIm[mt][nt]);
      }
    }
  }

  // C/D layout: VGPR j -> (M = j + 8*half, N = l16) within each 16x16 tile
#pragma unroll
  for (int mt = 0; mt < MT; ++mt) {
    const int mbase = (mt0 + mt) * 16 + half * 8;
#pragma unroll
    for (int nt = 0; nt < 4; ++nt) {
      const int n = nbase + nt * 16 + l16;
      const float2 bv = bias[n];
#pragma unroll
      for (int j = 0; j < 8; ++j) {
        float2 o;
        o.x = accRe[mt][nt][j] + bv.x;
        o.y = accIm[mt][nt][j] + bv.y;
        out[(size_t)(mbase + j) * N + n] = o;
      }
    }
  }
}

// ---------------------------------------------------------------------------
// Kernel: projection h(512) -> c(8) -> m(4) -> q,k,v(4).  One wave per row.
// ---------------------------------------------------------------------------
__global__ __launch_bounds__(256)
void qm_proj(const float2* __restrict__ h,
             const float2* __restrict__ Wcoh, const float2* __restrict__ bcoh,
             const float2* __restrict__ Wmot, const float2* __restrict__ bmot,
             const float2* __restrict__ Wq, const float2* __restrict__ bq,
             const float2* __restrict__ Wk, const float2* __restrict__ bk,
             const float2* __restrict__ Wv, const float2* __restrict__ bv,
             float2* __restrict__ qo, float2* __restrict__ ko,
             float2* __restrict__ vo) {
  const int wave = threadIdx.x >> 5;
  const int lane = threadIdx.x & 31;
  const int row  = blockIdx.x * 8 + wave;
  const float2* hrow = h + (size_t)row * Dh;

  float cre[Ch], cim[Ch];
#pragma unroll
  for (int j = 0; j < Ch; ++j) { cre[j] = 0.f; cim[j] = 0.f; }

  for (int k = lane; k < Dh; k += 32) {
    float2 hv = hrow[k];
#pragma unroll
    for (int j = 0; j < Ch; ++j) {
      float2 w = Wcoh[j * Dh + k];
      cre[j] = fmaf(hv.x, w.x, fmaf(-hv.y, w.y, cre[j]));
      cim[j] = fmaf(hv.x, w.y, fmaf(hv.y, w.x, cim[j]));
    }
  }
#pragma unroll
  for (int off = 16; off >= 1; off >>= 1) {
#pragma unroll
    for (int j = 0; j < Ch; ++j) {
      cre[j] += __shfl_xor(cre[j], off, 32);
      cim[j] += __shfl_xor(cim[j], off, 32);
    }
  }

  if (lane == 0) {
#pragma unroll
    for (int j = 0; j < Ch; ++j) { cre[j] += bcoh[j].x; cim[j] += bcoh[j].y; }
    float mre[Rk], mim[Rk];
#pragma unroll
    for (int i = 0; i < Rk; ++i) {
      mre[i] = bmot[i].x; mim[i] = bmot[i].y;
#pragma unroll
      for (int j = 0; j < Ch; ++j) {
        float2 w = Wmot[i * Ch + j];
        mre[i] += cre[j] * w.x - cim[j] * w.y;
        mim[i] += cre[j] * w.y + cim[j] * w.x;
      }
    }
#pragma unroll
    for (int i = 0; i < Rk; ++i) {
      float2 aq = bq[i], ak = bk[i], av = bv[i];
#pragma unroll
      for (int r = 0; r < Rk; ++r) {
        float2 wq = Wq[i * Rk + r], wk = Wk[i * Rk + r], wv = Wv[i * Rk + r];
        aq.x += mre[r] * wq.x - mim[r] * wq.y;  aq.y += mre[r] * wq.y + mim[r] * wq.x;
        ak.x += mre[r] * wk.x - mim[r] * wk.y;  ak.y += mre[r] * wk.y + mim[r] * wk.x;
        av.x += mre[r] * wv.x - mim[r] * wv.y;  av.y += mre[r] * wv.y + mim[r] * wv.x;
      }
      qo[(size_t)row * Rk + i] = aq;
      ko[(size_t)row * Rk + i] = ak;
      vo[(size_t)row * Rk + i] = av;
    }
  }
}

// ---------------------------------------------------------------------------
// Kernel: flash-style attention.  softmax over Re(q . k), applied to complex v.
// One thread per query row; k/v chunks staged in LDS with async copies.
// Grid: (Sq/256, Bz), block 256.
// ---------------------------------------------------------------------------
constexpr int ATT_CHUNK = 512;

__global__ __launch_bounds__(256)
void qm_attn(const float2* __restrict__ q, const float2* __restrict__ k,
             const float2* __restrict__ v, float2* __restrict__ o) {
  __shared__ float2 sk[ATT_CHUNK * Rk];
  __shared__ float2 sv[ATT_CHUNK * Rk];

  const int tid = threadIdx.x;
  const int b = blockIdx.y;
  const int s = blockIdx.x * 256 + tid;
  const float2* qrow = q + ((size_t)b * Sq + s) * Rk;

  float qre[Rk], qim[Rk];
#pragma unroll
  for (int r = 0; r < Rk; ++r) { float2 t = qrow[r]; qre[r] = t.x; qim[r] = t.y; }

  float mmax = -3.0e38f, lsum = 0.f;
  float are[Rk] = {0.f, 0.f, 0.f, 0.f};
  float aim[Rk] = {0.f, 0.f, 0.f, 0.f};

  for (int c0 = 0; c0 < Sq; c0 += ATT_CHUNK) {
    __syncthreads();
    const float2* kg = k + ((size_t)b * Sq + c0) * Rk;
    const float2* vg = v + ((size_t)b * Sq + c0) * Rk;
    // 16 KB per array = 1024 b128 transfers; 4 per thread per array
#pragma unroll
    for (int j = 0; j < 4; ++j) {
      const int i = tid + j * 256;                 // 16-byte transfer index
      const unsigned int go = (unsigned int)(i * 16);
      qm_async_b128(qm_lds_off(&sk[i * 2]), go, kg);
      qm_async_b128(qm_lds_off(&sv[i * 2]), go, vg);
    }
    qm_wait_async0();
    __syncthreads();

    for (int t = 0; t < ATT_CHUNK; ++t) {
      float sc = 0.f;
#pragma unroll
      for (int r = 0; r < Rk; ++r) {
        float2 kv = sk[t * Rk + r];
        sc += qre[r] * kv.x - qim[r] * kv.y;       // Re(q*k), no conjugation
      }
      sc *= ATT_SCALE;
      float mn = fmaxf(mmax, sc);
      float corr = __expf(mmax - mn);
      float p = __expf(sc - mn);
      lsum = lsum * corr + p;
#pragma unroll
      for (int r = 0; r < Rk; ++r) {
        float2 vv = sv[t * Rk + r];
        are[r] = are[r] * corr + p * vv.x;
        aim[r] = aim[r] * corr + p * vv.y;
      }
      mmax = mn;
    }
  }

  const float inv = 1.f / lsum;
#pragma unroll
  for (int r = 0; r < Rk; ++r) {
    float2 t; t.x = are[r] * inv; t.y = aim[r] * inv;
    o[((size_t)b * Sq + s) * Rk + r] = t;
  }
}

// ---------------------------------------------------------------------------
// Kernel: back-projection o(4) -> c2(8) -> h2(512) as bf16 planes.
// ---------------------------------------------------------------------------
__global__ __launch_bounds__(256)
void qm_backproj(const float2* __restrict__ o,
                 const float2* __restrict__ Wmi, const float2* __restrict__ bmi,
                 const float2* __restrict__ Wci, const float2* __restrict__ bci,
                 bf16_t* __restrict__ h2r, bf16_t* __restrict__ h2i) {
  const int wave = threadIdx.x >> 5;
  const int lane = threadIdx.x & 31;
  const int row  = blockIdx.x * 8 + wave;

  float ore[Rk], oim[Rk];
#pragma unroll
  for (int r = 0; r < Rk; ++r) {
    float2 t = o[(size_t)row * Rk + r];
    ore[r] = t.x; oim[r] = t.y;
  }
  float c2re[Ch], c2im[Ch];
#pragma unroll
  for (int j = 0; j < Ch; ++j) {
    c2re[j] = bmi[j].x; c2im[j] = bmi[j].y;
#pragma unroll
    for (int r = 0; r < Rk; ++r) {
      float2 w = Wmi[j * Rk + r];
      c2re[j] += ore[r] * w.x - oim[r] * w.y;
      c2im[j] += ore[r] * w.y + oim[r] * w.x;
    }
  }
#pragma unroll 4
  for (int t = 0; t < Dh / 32; ++t) {
    const int n = t * 32 + lane;
    float re = bci[n].x, im = bci[n].y;
#pragma unroll
    for (int j = 0; j < Ch; ++j) {
      float2 w = Wci[n * Ch + j];
      re += c2re[j] * w.x - c2im[j] * w.y;
      im += c2re[j] * w.y + c2im[j] * w.x;
    }
    h2r[(size_t)row * Dh + n] = (bf16_t)re;
    h2i[(size_t)row * Dh + n] = (bf16_t)im;
  }
}

// ---------------------------------------------------------------------------
// Host launcher
// ---------------------------------------------------------------------------
extern "C" void kernel_launch(void* const* d_in, const int* in_sizes, int n_in,
                              void* d_out, int out_size, void* d_ws, size_t ws_size,
                              hipStream_t stream) {
  const float2* x        = (const float2*)d_in[0];
  const float2* W_in     = (const float2*)d_in[1];
  const float2* b_in     = (const float2*)d_in[2];
  const float2* W_coh    = (const float2*)d_in[3];
  const float2* b_coh    = (const float2*)d_in[4];
  const float2* W_mot    = (const float2*)d_in[5];
  const float2* b_mot    = (const float2*)d_in[6];
  const float2* W_q      = (const float2*)d_in[7];
  const float2* b_q      = (const float2*)d_in[8];
  const float2* W_k      = (const float2*)d_in[9];
  const float2* b_k      = (const float2*)d_in[10];
  const float2* W_v      = (const float2*)d_in[11];
  const float2* b_v      = (const float2*)d_in[12];
  const float2* W_moti   = (const float2*)d_in[13];
  const float2* b_moti   = (const float2*)d_in[14];
  const float2* W_cohi   = (const float2*)d_in[15];
  const float2* b_cohi   = (const float2*)d_in[16];
  const float2* W_out    = (const float2*)d_in[17];
  const float2* b_out    = (const float2*)d_in[18];
  float2* out = (float2*)d_out;

  char* ws = (char*)d_ws;
  size_t off = 0;
  auto carve = [&](size_t bytes) -> void* {
    void* p = ws + off;
    off += (bytes + 255) & ~(size_t)255;
    return p;
  };
  const size_t MD = (size_t)Mrows * Dh;
  const size_t DD = (size_t)Dh * Dh;
  bf16_t* xr   = (bf16_t*)carve(MD * 2);          // reused as h2r later
  bf16_t* xi   = (bf16_t*)carve(MD * 2);          // reused as h2i later
  bf16_t* wr1  = (bf16_t*)carve(DD * 2);
  bf16_t* wi1  = (bf16_t*)carve(DD * 2);
  bf16_t* wn1  = (bf16_t*)carve(DD * 2);
  bf16_t* wr2  = (bf16_t*)carve(DD * 2);
  bf16_t* wi2  = (bf16_t*)carve(DD * 2);
  bf16_t* wn2  = (bf16_t*)carve(DD * 2);
  float2* h    = (float2*)carve(MD * sizeof(float2));
  float2* qb   = (float2*)carve((size_t)Mrows * Rk * sizeof(float2));
  float2* kb   = (float2*)carve((size_t)Mrows * Rk * sizeof(float2));
  float2* vb   = (float2*)carve((size_t)Mrows * Rk * sizeof(float2));
  float2* ob   = (float2*)carve((size_t)Mrows * Rk * sizeof(float2));
  (void)ws_size; (void)in_sizes; (void)n_in; (void)out_size;

  // 1) bf16 planes
  qm_split_x<<<dim3((MD + 255) / 256), 256, 0, stream>>>(x, xr, xi, (int)MD);
  qm_split_w<<<dim3((DD + 255) / 256), 256, 0, stream>>>(W_in, wr1, wi1, wn1, (int)DD);
  qm_split_w<<<dim3((DD + 255) / 256), 256, 0, stream>>>(W_out, wr2, wi2, wn2, (int)DD);

  // 2) h = x @ W_in^T + b_in
  qm_gemm_cplx<<<dim3(Mrows / 256, Dh / 64), 256, 0, stream>>>(
      xr, xi, wr1, wi1, wn1, b_in, h);

  // 3) h -> c -> m -> q,k,v
  qm_proj<<<dim3(Mrows / 8), 256, 0, stream>>>(
      h, W_coh, b_coh, W_mot, b_mot, W_q, b_q, W_k, b_k, W_v, b_v, qb, kb, vb);

  // 4) flash attention
  qm_attn<<<dim3(Sq / 256, Bz), 256, 0, stream>>>(qb, kb, vb, ob);

  // 5) o -> c2 -> h2 (bf16 planes, reuse xr/xi)
  qm_backproj<<<dim3(Mrows / 8), 256, 0, stream>>>(
      ob, W_moti, b_moti, W_cohi, b_cohi, xr, xi);

  // 6) out = h2 @ W_out^T + b_out
  qm_gemm_cplx<<<dim3(Mrows / 256, Dh / 64), 256, 0, stream>>>(
      xr, xi, wr2, wi2, wn2, b_out, out);
}
